// GTransCustomModular_80711025426696
// MI455X (gfx1250) — matrix operand order repaired and checked by reading, not compiled, not run on hardware
//
#include <hip/hip_runtime.h>
#include <hip/hip_bf16.h>
#include <math.h>

// ---------------------------------------------------------------------------
// GTransCustomModular (graph transformer, 2x TransformerConv + final conv)
// MI455X / gfx1250: wave32, WMMA 16x16x32 f16->f32 for all dense GEMMs.
// GEMM is specialized on K so the hot K=128 path stages tiles with
// unconditional coalesced global_load_b128 (no exec-mask predication),
// one barrier, then a fully unrolled chain of ds_load_b128 -> v_wmma.
// Edge phase: q/k/v (77MB) are L2-resident on the 192MB L2 -> gathers cheap.
// ---------------------------------------------------------------------------

typedef __attribute__((ext_vector_type(16))) _Float16 v16h;
typedef __attribute__((ext_vector_type(8)))  _Float16 v8h;
typedef __attribute__((ext_vector_type(4)))  _Float16 v4h;
typedef __attribute__((ext_vector_type(8)))  float    v8f;
typedef __attribute__((ext_vector_type(4)))  float    f4v;

#define HEADS 4
constexpr int NN  = 50000;            // nodes
constexpr int NE  = 600000;           // edges
constexpr int ND  = 128;              // node feature dim
constexpr int ED  = 16;               // edge feature dim (initial)
constexpr int ED2 = ED + HEADS;       // edge dim after alpha concat = 20
constexpr float SCALE32 = 0.17677669529663687f;   // 1/sqrt(32)

// ---- monotone float<->uint encoding so we can do float-max with atomicMax(u32)
__device__ __forceinline__ unsigned fenc(float f) {
  unsigned b = __float_as_uint(f);
  return b ^ ((b & 0x80000000u) ? 0xFFFFFFFFu : 0x80000000u);
}
__device__ __forceinline__ float fdec(unsigned e) {
  unsigned b = e ^ ((e & 0x80000000u) ? 0x80000000u : 0xFFFFFFFFu);
  return __uint_as_float(b);
}
#define FENC_NEG_INF 0x007FFFFFu   // fenc(-inf)

__global__ __launch_bounds__(256)
void fill_u32_kernel(unsigned* __restrict__ p, unsigned v, int n) {
  int i = blockIdx.x * 256 + threadIdx.x;
  if (i < n) p[i] = v;
}

// ---------------------------------------------------------------------------
// WMMA GEMM:  C[M,Nout] = A[M,K] @ W[K,Nout] + bias     (M % 16 == 0 here)
// Block: 256 threads / 8 waves -> 128x16 C tile; one 16x16 WMMA tile per wave.
// Whole K staged to LDS once, single barrier, then KCEIL/32 chained WMMAs.
// Fragment layouts per CDNA5 ISA 7.12.2 (wave32):
//   A (16x32 f16): row m = lane%16, elem j -> K = (j&7) + ((j>>3)<<4) + (lane>>4)*8
//   B (32x16 f16): col n = lane%16, elem j -> K = (j&7) + ((j>>3)<<4) + (lane>>4)*8
//   C (16x16 f32): col n = lane%16, vgpr r -> M = r + (lane>>4)*8
// ---------------------------------------------------------------------------
template <int K>
__global__ __launch_bounds__(256)
void wmma_gemm_kernel(const float* __restrict__ A, const float* __restrict__ W,
                      const float* __restrict__ bias, float* __restrict__ C,
                      int M, int Nout) {
  constexpr int KC = (K + 31) & ~31;     // K rounded to WMMA depth
  constexpr int KP = KC + 8;             // LDS row stride in halfs:
                                         //  KP%8==0 -> 16B-aligned rows
  __shared__ __align__(16) _Float16 lA[128 * KP];
  __shared__ __align__(16) _Float16 lB[16 * KP];

  const int t     = threadIdx.x;
  const int lane  = t & 31;
  const int wave  = t >> 5;                 // 0..7
  const int mBase = blockIdx.x * 128;
  const int nTile = blockIdx.y * 16;
  const int hi    = lane >> 4;
  const int l15   = lane & 15;
  const int nCol  = nTile + l15;
  const int mTile = mBase + wave * 16;

  // ---- stage A (128 x KC) as f16; chunked so 4 b128 loads stay in flight ----
  constexpr int ITA = (128 * (KC / 4)) / 256;     // float4 slots per thread
#pragma unroll
  for (int g = 0; g < ITA; g += 4) {
    f4v v[4];
#pragma unroll
    for (int u = 0; u < 4; ++u) {
      int lin = t + (g + u) * 256;
      int row = lin / (KC / 4);
      int kq  = (lin % (KC / 4)) * 4;
      int rg  = mBase + row; if (rg >= M) rg = M - 1;   // clamp: value unused
      if constexpr (K == KC) {
        v[u] = *(const f4v*)(A + (size_t)rg * K + kq);  // unconditional
      } else {
        f4v z = {0.f, 0.f, 0.f, 0.f};
        v[u] = (kq < K) ? *(const f4v*)(A + (size_t)rg * K + kq) : z;
      }
    }
#pragma unroll
    for (int u = 0; u < 4; ++u) {
      int lin = t + (g + u) * 256;
      int row = lin / (KC / 4);
      int kq  = (lin % (KC / 4)) * 4;
      v4h h; h[0] = (_Float16)v[u][0]; h[1] = (_Float16)v[u][1];
             h[2] = (_Float16)v[u][2]; h[3] = (_Float16)v[u][3];
      *(v4h*)(lA + row * KP + kq) = h;
    }
  }
  // ---- stage W (KC x 16), transposed: lB[n][kb] ----
  constexpr int ITB = (KC * 16) / 256;
#pragma unroll
  for (int i = 0; i < ITB; ++i) {
    int lin = t + i * 256;
    int kb  = lin >> 4;
    int n   = lin & 15;
    int ng  = nTile + n;
    float v = 0.f;
    if constexpr (K == KC) {
      if (ng < Nout) v = W[(size_t)kb * Nout + ng];
    } else {
      if (kb < K && ng < Nout) v = W[(size_t)kb * Nout + ng];
    }
    lB[n * KP + kb] = (_Float16)v;
  }
  __syncthreads();

  // ---- unrolled WMMA chain over K ----
  const _Float16* la = lA + (wave * 16 + l15) * KP;
  const _Float16* lb = lB + l15 * KP;
  v8f acc = {};
#pragma unroll
  for (int kk = 0; kk < KC; kk += 32) {
    v8h alo = *(const v8h*)(la + kk + hi * 8);        // K = kk+hi*8 .. +7
    v8h ahi = *(const v8h*)(la + kk + 16 + hi * 8);   // K = kk+16+hi*8 .. +7
    v8h blo = *(const v8h*)(lb + kk + hi * 8);        // K = kk+hi*8 .. +7
    v8h bhi = *(const v8h*)(lb + kk + 16 + hi * 8);   // K = kk+16+hi*8 .. +7
    v16h a = __builtin_shufflevector(alo, ahi, 0,1,2,3,4,5,6,7,8,9,10,11,12,13,14,15);
    v16h b = __builtin_shufflevector(blo, bhi, 0,1,2,3,4,5,6,7,8,9,10,11,12,13,14,15);
    acc = __builtin_amdgcn_wmma_f32_16x16x32_f16(
        /*neg_a=*/false, a, /*neg_b=*/false, b,
        /*c_mod=*/(short)0, acc, /*reuse_a=*/false, /*reuse_b=*/false);
  }

  if (mTile < M && nCol < Nout) {       // M % 16 == 0 -> all 16 rows valid
    float bb = (bias != nullptr) ? bias[nCol] : 0.0f;
#pragma unroll
    for (int r = 0; r < 8; ++r) {
      int m = mTile + r + hi * 8;
      C[(size_t)m * Nout + nCol] = acc[r] + bb;
    }
  }
}

// ---------------------------------------------------------------------------
// Edge pass 1: alpha_raw[e,h] = <q[dst,h,:], k[src,h,:] + e[e,h,:]> * scale
//              segment-max into mmax[dst,h] (float via monotone-uint atomicMax)
// ---------------------------------------------------------------------------
template <int C>
__global__ __launch_bounds__(256)
void edge_alpha_kernel(const int* __restrict__ src, const int* __restrict__ dst,
                       const float* __restrict__ q, const float* __restrict__ k,
                       const float* __restrict__ e, float* __restrict__ araw,
                       unsigned* __restrict__ mmax, int nE, float scale) {
  int idx = blockIdx.x * 256 + threadIdx.x;
  if (idx >= nE * HEADS) return;
  int eid = idx >> 2, h = idx & 3;                    // HEADS == 4
  int s = src[eid], d = dst[eid];
  constexpr int HC = HEADS * C;
  const float* qp = q + (size_t)d * HC + h * C;
  const float* kp = k + (size_t)s * HC + h * C;
  const float* ep = e + (size_t)eid * HC + h * C;
  float acc = 0.0f;
  if constexpr ((C & 3) == 0) {
    const f4v* q4 = (const f4v*)qp;
    const f4v* k4 = (const f4v*)kp;
    const f4v* e4 = (const f4v*)ep;
#pragma unroll
    for (int c = 0; c < C / 4; ++c) {
      f4v qv = q4[c], kv = k4[c], ev = e4[c];
      acc = fmaf(qv[0], kv[0] + ev[0], acc);
      acc = fmaf(qv[1], kv[1] + ev[1], acc);
      acc = fmaf(qv[2], kv[2] + ev[2], acc);
      acc = fmaf(qv[3], kv[3] + ev[3], acc);
    }
  } else {
    for (int c = 0; c < C; ++c) acc = fmaf(qp[c], kp[c] + ep[c], acc);
  }
  acc *= scale;
  araw[idx] = acc;
  atomicMax(&mmax[d * HEADS + h], fenc(acc));
}

// Edge pass 2: ex = exp(alpha_raw - mmax[dst,h]); segment-sum into ssum[dst,h]
__global__ __launch_bounds__(256)
void edge_expsum_kernel(const int* __restrict__ dst, float* __restrict__ aex,
                        const unsigned* __restrict__ mmax,
                        float* __restrict__ ssum, int nE) {
  int idx = blockIdx.x * 256 + threadIdx.x;
  if (idx >= nE * HEADS) return;
  int eid = idx >> 2, h = idx & 3;
  int d = dst[eid];
  float m  = fdec(mmax[d * HEADS + h]);
  float ex = __expf(aex[idx] - m);
  aex[idx] = ex;
  atomicAdd(&ssum[d * HEADS + h], ex);
}

// Edge pass 3: alpha = ex / (sum + 1e-16); scatter (v[src]+e)*alpha into agg[dst]
template <int C>
__global__ __launch_bounds__(256)
void edge_message_kernel(const int* __restrict__ src, const int* __restrict__ dst,
                         const float* __restrict__ v, const float* __restrict__ e,
                         const float* __restrict__ aex, const float* __restrict__ ssum,
                         float* __restrict__ alpha, float* __restrict__ agg,
                         int nE) {
  int idx = blockIdx.x * 256 + threadIdx.x;
  if (idx >= nE * HEADS) return;
  int eid = idx >> 2, h = idx & 3;
  int s = src[eid], d = dst[eid];
  float a = aex[idx] / (ssum[d * HEADS + h] + 1e-16f);
  alpha[idx] = a;
  constexpr int HC = HEADS * C;
  const float* vp = v + (size_t)s * HC + h * C;
  const float* ep = e + (size_t)eid * HC + h * C;
  float* op = agg + (size_t)d * HC + h * C;
  if constexpr ((C & 3) == 0) {
    const f4v* v4 = (const f4v*)vp;
    const f4v* e4 = (const f4v*)ep;
#pragma unroll
    for (int c = 0; c < C / 4; ++c) {
      f4v vv = v4[c], ev = e4[c];
      atomicAdd(&op[c * 4 + 0], (vv[0] + ev[0]) * a);
      atomicAdd(&op[c * 4 + 1], (vv[1] + ev[1]) * a);
      atomicAdd(&op[c * 4 + 2], (vv[2] + ev[2]) * a);
      atomicAdd(&op[c * 4 + 3], (vv[3] + ev[3]) * a);
    }
  } else {
    for (int c = 0; c < C; ++c) atomicAdd(&op[c], (vp[c] + ep[c]) * a);
  }
}

// ---------------------------------------------------------------------------
// Beta-gate + LayerNorm, one wave32 per node (HC == 128, 4 elems/lane).
//   g = sigmoid(betaW . [out, x_r, out - x_r]);  y = g*x_r + (1-g)*out
//   x' = LN(y) * gamma + beta
// ---------------------------------------------------------------------------
__global__ __launch_bounds__(256)
void gate_ln_kernel(const float* __restrict__ agg, const float* __restrict__ skip,
                    const float* __restrict__ betaW, const float* __restrict__ gln,
                    const float* __restrict__ bln, float* __restrict__ xout, int n) {
  const int HC   = 128;
  int node = blockIdx.x * 8 + (threadIdx.x >> 5);
  if (node >= n) return;
  int lane = threadIdx.x & 31;

  float o[4], r[4];
  float dotv = 0.0f;
#pragma unroll
  for (int i = 0; i < 4; ++i) {
    int j = lane + i * 32;
    o[i] = agg[(size_t)node * HC + j];
    r[i] = skip[(size_t)node * HC + j];
    dotv += o[i] * betaW[j] + r[i] * betaW[HC + j] + (o[i] - r[i]) * betaW[2 * HC + j];
  }
#pragma unroll
  for (int off = 16; off > 0; off >>= 1) dotv += __shfl_xor(dotv, off, 32);
  float g = 1.0f / (1.0f + __expf(-dotv));

  float y[4], mu = 0.0f;
#pragma unroll
  for (int i = 0; i < 4; ++i) { y[i] = g * r[i] + (1.0f - g) * o[i]; mu += y[i]; }
#pragma unroll
  for (int off = 16; off > 0; off >>= 1) mu += __shfl_xor(mu, off, 32);
  mu *= (1.0f / HC);

  float var = 0.0f;
#pragma unroll
  for (int i = 0; i < 4; ++i) { float d2 = y[i] - mu; var += d2 * d2; }
#pragma unroll
  for (int off = 16; off > 0; off >>= 1) var += __shfl_xor(var, off, 32);
  var *= (1.0f / HC);
  float inv = rsqrtf(var + 1e-5f);

#pragma unroll
  for (int i = 0; i < 4; ++i) {
    int j = lane + i * 32;
    xout[(size_t)node * HC + j] = (y[i] - mu) * inv * gln[j] + bln[j];
  }
}

// edge_attr2[e, 0:16] = edge_attr[e, :]; edge_attr2[e, 16:20] = alpha[e, :]
__global__ __launch_bounds__(256)
void concat_edge_kernel(const float* __restrict__ ea, const float* __restrict__ alpha,
                        float* __restrict__ ea2, int nE) {
  int i = blockIdx.x * 256 + threadIdx.x;
  int total = nE * ED2;
  if (i >= total) return;
  int eid = i / ED2, c = i - eid * ED2;
  ea2[i] = (c < ED) ? ea[(size_t)eid * ED + c] : alpha[eid * HEADS + (c - ED)];
}

// overwrite alpha slice only (base edge_attr already in place)
__global__ __launch_bounds__(256)
void update_alpha_kernel(const float* __restrict__ alpha, float* __restrict__ ea2, int nE) {
  int i = blockIdx.x * 256 + threadIdx.x;
  if (i >= nE * HEADS) return;
  int eid = i >> 2, h = i & 3;
  ea2[(size_t)eid * ED2 + ED + h] = alpha[i];
}

// Final conv epilogue: out = sigmoid(g*x_r + (1-g)*mean_h(agg)), c=1, concat=False
__global__ __launch_bounds__(256)
void final_out_kernel(const float* __restrict__ agg, const float* __restrict__ skip1,
                      const float* __restrict__ bW, float* __restrict__ out, int n) {
  int i = blockIdx.x * 256 + threadIdx.x;
  if (i >= n) return;
  float o = 0.25f * (agg[i * 4 + 0] + agg[i * 4 + 1] + agg[i * 4 + 2] + agg[i * 4 + 3]);
  float xr = skip1[i];
  float t = bW[0] * o + bW[1] * xr + bW[2] * (o - xr);
  float g = 1.0f / (1.0f + __expf(-t));
  float y = g * xr + (1.0f - g) * o;
  out[i] = 1.0f / (1.0f + __expf(-y));
}

// ---------------------------------------------------------------------------
extern "C" void kernel_launch(void* const* d_in, const int* in_sizes, int n_in,
                              void* d_out, int out_size, void* d_ws, size_t ws_size,
                              hipStream_t stream) {
  (void)in_sizes; (void)n_in; (void)out_size; (void)ws_size;

  const float* x   = (const float*)d_in[0];
  const int*   src = (const int*)d_in[1];
  const int*   dst = src + NE;
  const float* ea0 = (const float*)d_in[2];
  auto P = [&](int i) { return (const float*)d_in[i]; };
  // conv0: 3..13  conv1: 14..24  ln: 25..28  final: 29..39
  const int C0 = 3, C1 = 14, LN = 25, FC = 29;
  // per-conv offsets: qW qb kW kb vW vb eW eb sW sb bW -> 0..10

  // ---- workspace carve ----
  char* wp = (char*)d_ws;
  auto alloc = [&](size_t bytes) -> void* {
    void* p = (void*)wp; wp += (bytes + 255) & ~(size_t)255; return p;
  };
  float*    qbuf = (float*)alloc((size_t)NN * ND * 4);
  float*    kbuf = (float*)alloc((size_t)NN * ND * 4);
  float*    vbuf = (float*)alloc((size_t)NN * ND * 4);
  float*    sbuf = (float*)alloc((size_t)NN * ND * 4);
  float*    ebuf = (float*)alloc((size_t)NE * ND * 4);
  float*    araw = (float*)alloc((size_t)NE * HEADS * 4);
  float*    alph = (float*)alloc((size_t)NE * HEADS * 4);
  unsigned* mmax = (unsigned*)alloc((size_t)NN * HEADS * 4);
  float*    ssum = (float*)alloc((size_t)NN * HEADS * 4);
  float*    agg  = (float*)alloc((size_t)NN * ND * 4);
  float*    x1   = (float*)alloc((size_t)NN * ND * 4);
  float*    x2   = (float*)alloc((size_t)NN * ND * 4);
  float*    ea2  = (float*)alloc((size_t)NE * ED2 * 4);

  auto gemm = [&](const float* A, const float* W, const float* b, float* Cd,
                  int M, int K, int Nout) {
    dim3 grid((M + 127) / 128, (Nout + 15) / 16);
    if (K == 128) {
      wmma_gemm_kernel<128><<<grid, 256, 0, stream>>>(A, W, b, Cd, M, Nout);
    } else if (K == 16) {
      wmma_gemm_kernel<16><<<grid, 256, 0, stream>>>(A, W, b, Cd, M, Nout);
    } else {
      wmma_gemm_kernel<20><<<grid, 256, 0, stream>>>(A, W, b, Cd, M, Nout);
    }
  };
  auto fill = [&](void* p, unsigned v, int n) {
    fill_u32_kernel<<<(n + 255) / 256, 256, 0, stream>>>((unsigned*)p, v, n);
  };
  const int EHB = (NE * HEADS + 255) / 256;

  auto conv = [&](const float* xin, int fin, const float* ea, int edim,
                  int base, int C, float scale, float* xln_out,
                  const float* gln, const float* bln) {
    int HC = HEADS * C;
    gemm(xin, P(base + 0), P(base + 1), qbuf, NN, fin, HC);
    gemm(xin, P(base + 2), P(base + 3), kbuf, NN, fin, HC);
    gemm(xin, P(base + 4), P(base + 5), vbuf, NN, fin, HC);
    gemm(xin, P(base + 8), P(base + 9), sbuf, NN, fin, (xln_out ? HC : 1));
    gemm(ea,  P(base + 6), P(base + 7), ebuf, NE, edim, HC);
    fill(mmax, FENC_NEG_INF, NN * HEADS);
    fill(ssum, 0u, NN * HEADS);
    fill(agg,  0u, NN * HC);
    if (C == 32) {
      edge_alpha_kernel<32><<<EHB, 256, 0, stream>>>(src, dst, qbuf, kbuf, ebuf,
                                                     araw, mmax, NE, scale);
    } else {
      edge_alpha_kernel<1><<<EHB, 256, 0, stream>>>(src, dst, qbuf, kbuf, ebuf,
                                                    araw, mmax, NE, scale);
    }
    edge_expsum_kernel<<<EHB, 256, 0, stream>>>(dst, araw, mmax, ssum, NE);
    if (C == 32) {
      edge_message_kernel<32><<<EHB, 256, 0, stream>>>(src, dst, vbuf, ebuf, araw,
                                                       ssum, alph, agg, NE);
    } else {
      edge_message_kernel<1><<<EHB, 256, 0, stream>>>(src, dst, vbuf, ebuf, araw,
                                                      ssum, alph, agg, NE);
    }
    if (xln_out) {
      gate_ln_kernel<<<(NN + 7) / 8, 256, 0, stream>>>(agg, sbuf, P(base + 10),
                                                       gln, bln, xln_out, NN);
    }
  };

  // ---- layer 0: x(N,128), edge_attr(E,16) -> x1, alpha0 ----
  conv(x, ND, ea0, ED, C0, 32, SCALE32, x1, P(LN + 0), P(LN + 1));
  concat_edge_kernel<<<(NE * ED2 + 255) / 256, 256, 0, stream>>>(ea0, alph, ea2, NE);

  // ---- layer 1: x1(N,128), ea2(E,20) -> x2, alpha1 ----
  conv(x1, ND, ea2, ED2, C1, 32, SCALE32, x2, P(LN + 2), P(LN + 3));
  update_alpha_kernel<<<EHB, 256, 0, stream>>>(alph, ea2, NE);

  // ---- final conv: heads=4, c=1, concat=False, then sigmoid ----
  conv(x2, ND, ea2, ED2, FC, 1, 1.0f, nullptr, nullptr, nullptr);
  final_out_kernel<<<(NN + 255) / 256, 256, 0, stream>>>(agg, sbuf, P(FC + 10),
                                                         (float*)d_out, NN);
}
